// TransformerModel_48876727828986
// MI455X (gfx1250) — compile-verified, matrix operand order checked by reading, not run on hardware
//
#include <hip/hip_runtime.h>

// Model dims (compile-time)
#define H_    1024
#define NH_   16
#define HD_   64
#define S_    1024
#define B_    2
#define I_    4096
#define V_    32000
#define L_    6
#define MR_   (B_*S_)   // 2048 token rows

typedef __attribute__((ext_vector_type(16))) __bf16       v16bf;
typedef __attribute__((ext_vector_type(8)))  float        v8f;
typedef __attribute__((ext_vector_type(8)))  unsigned int v8u;
typedef unsigned short u16;
typedef unsigned int   u32;

__device__ __forceinline__ u32 f2bf1(float a){
    u32 u = __float_as_uint(a);
    u = (u + 0x7FFFu + ((u >> 16) & 1u)) >> 16;
    return u & 0xFFFFu;
}
__device__ __forceinline__ u32 pack_bf2(float a, float b){
    return f2bf1(a) | (f2bf1(b) << 16);
}
__device__ __forceinline__ v8f wmma_bf16(v16bf a, v16bf b, v8f c){
    return __builtin_amdgcn_wmma_f32_16x16x32_bf16(false, a, false, b, (short)0, c, false, false);
}
// 16-lane-striped bf16 fragment (8 u32 = 16 bf16 per lane); off = ks*16 + half*4
__device__ __forceinline__ v16bf frag_ld(const u32* base, int off){
    uint4 a0 = *(const uint4*)(base + off);
    uint4 a1 = *(const uint4*)(base + off + 8);
    v8u u = {a0.x, a0.y, a0.z, a0.w, a1.x, a1.y, a1.z, a1.w};
    return __builtin_bit_cast(v16bf, u);
}
__device__ __forceinline__ float gelu_f(float x){
    return 0.5f * x * (1.0f + erff(x * 0.70710678118654752f));
}
// 64B global -> LDS async copy (4 x b128), tracked by ASYNCcnt
__device__ __forceinline__ void async_copy64(u32 lds_byte, const void* gaddr){
    asm volatile(
        "global_load_async_to_lds_b128 %0, %1, off\n\t"
        "global_load_async_to_lds_b128 %0, %1, off offset:16\n\t"
        "global_load_async_to_lds_b128 %0, %1, off offset:32\n\t"
        "global_load_async_to_lds_b128 %0, %1, off offset:48"
        :: "v"(lds_byte), "v"(gaddr) : "memory");
}
__device__ __forceinline__ void wait_async0(){
    asm volatile("s_wait_asynccnt 0x0" ::: "memory");
}

// ---------------------------------------------------------------------------
// Embedding: h[row,:] = emb[ids[row],:] + pos[row%S,:]  (writes f32 + bf16)
// ---------------------------------------------------------------------------
__global__ __launch_bounds__(256) void embed_kernel(const int* __restrict__ ids,
                                                    const float* __restrict__ emb,
                                                    const float* __restrict__ pos,
                                                    float* __restrict__ h,
                                                    u16* __restrict__ hb){
    int gid = blockIdx.x * 256 + threadIdx.x;
    int col = gid & (H_ - 1);
    int row = gid >> 10;
    int s   = row & (S_ - 1);
    float v = emb[(size_t)ids[row] * H_ + col] + pos[(size_t)s * H_ + col];
    h[(size_t)gid]  = v;
    hb[(size_t)gid] = (u16)f2bf1(v);
}

// ---------------------------------------------------------------------------
// Weight transpose+convert: W[K][N] f32 -> Wt[N][K] bf16 (tiled via LDS)
// grid = (K/64, N/64), block = 256
// ---------------------------------------------------------------------------
__global__ __launch_bounds__(256) void wt_kernel(const float* __restrict__ W,
                                                 u16* __restrict__ Wt,
                                                 int K, int N){
    __shared__ __align__(16) u16 sT[64 * 68];
    const int t  = threadIdx.x;
    const int k0 = blockIdx.x * 64, n0 = blockIdx.y * 64;
    {
        const int kl = t >> 2, ng = (t & 3) * 16;
        const float* wp = W + (size_t)(k0 + kl) * N + n0 + ng;
        u32* sTu = (u32*)sT;
        #pragma unroll
        for (int j = 0; j < 4; ++j){
            float4 f = ((const float4*)wp)[j];
            sTu[kl * 34 + (ng >> 1) + j * 2]     = pack_bf2(f.x, f.y);
            sTu[kl * 34 + (ng >> 1) + j * 2 + 1] = pack_bf2(f.z, f.w);
        }
    }
    __syncthreads();
    {
        const int nl = t >> 2, kg = (t & 3) * 16;
        u16 tmp[16];
        #pragma unroll
        for (int j = 0; j < 16; ++j) tmp[j] = sT[(kg + j) * 68 + nl];
        u16* op = Wt + (size_t)(n0 + nl) * K + k0 + kg;
        uint4 o0, o1;
        o0.x = tmp[0]  | (tmp[1]  << 16); o0.y = tmp[2]  | (tmp[3]  << 16);
        o0.z = tmp[4]  | (tmp[5]  << 16); o0.w = tmp[6]  | (tmp[7]  << 16);
        o1.x = tmp[8]  | (tmp[9]  << 16); o1.y = tmp[10] | (tmp[11] << 16);
        o1.z = tmp[12] | (tmp[13] << 16); o1.w = tmp[14] | (tmp[15] << 16);
        ((uint4*)op)[0] = o0;
        ((uint4*)op)[1] = o1;
    }
}

// ---------------------------------------------------------------------------
// GEMM: C[M,N] = act(A[M,K]bf16 @ Wt[N,K]bf16^T + bias) * scale
// block 256 (8 waves as 2x4), block tile 128x256, wave tile 64x64,
// K-step 64, async global->LDS double-buffer, ONE barrier per K-step,
// 32 WMMA per K-step at ~1:1 frag-load:wmma ratio.
// OUT: 0 = f32 [M][N], 1 = bf16 [M][N], 2 = bf16 V-transposed [(b*H+col)][S]
// ---------------------------------------------------------------------------
#define TSTR 36   // u32 stride per LDS tile row (72 bf16 = 144B, 16B-aligned)

template<int ACT, int HAS_BIAS, int OUT>
__global__ __launch_bounds__(256) void gemm_kernel(const u16* __restrict__ A,
                                                   const u16* __restrict__ Wt,
                                                   const float* __restrict__ bias,
                                                   void* __restrict__ Cv,
                                                   int M, int N, int K, float scale){
    __shared__ __align__(16) u32 sA[2][128 * TSTR];
    __shared__ __align__(16) u32 sB[2][256 * TSTR];

    const int tid    = threadIdx.x;
    const int wave   = tid >> 5, lane = tid & 31;
    const int half   = lane >> 4, l15 = lane & 15;
    const int wave_m = wave >> 2;          // 0..1 -> 64-row slice
    const int wave_n = wave & 3;           // 0..3 -> 64-col slice
    const int bm     = blockIdx.y * 128, bn = blockIdx.x * 256;

    // staging: A = 1x64B/thread, B = 2x64B/thread per tile
    const int row = tid >> 1;              // 0..127
    const int seg = tid & 1;               // 32-bf16 half of a 64-wide row
    const u16* gA  = A  + (size_t)(bm + row) * K + seg * 32;
    const u16* gB0 = Wt + (size_t)(bn + row) * K + seg * 32;
    const u16* gB1 = Wt + (size_t)(bn + 128 + row) * K + seg * 32;
    u32 ldsA[2], ldsB0[2], ldsB1[2];
    #pragma unroll
    for (int p = 0; p < 2; ++p){
        ldsA[p]  = (u32)(size_t)&sA[p][row * TSTR + seg * 16];
        ldsB0[p] = (u32)(size_t)&sB[p][row * TSTR + seg * 16];
        ldsB1[p] = (u32)(size_t)&sB[p][(128 + row) * TSTR + seg * 16];
    }

    v8f acc[4][4] = {};
    const int NT = K >> 6;

    async_copy64(ldsA[0],  gA);
    async_copy64(ldsB0[0], gB0);
    async_copy64(ldsB1[0], gB1);

    for (int t = 0; t < NT; ++t){
        wait_async0();                     // this wave's slice of tile t landed
        __syncthreads();                   // everyone's slice landed
        if (t + 1 < NT){
            const size_t ko = (size_t)(t + 1) * 64;
            async_copy64(ldsA[(t + 1) & 1],  gA  + ko);
            async_copy64(ldsB0[(t + 1) & 1], gB0 + ko);
            async_copy64(ldsB1[(t + 1) & 1], gB1 + ko);
        }
        const u32* bufA = sA[t & 1];
        const u32* bufB = sB[t & 1];
        #pragma unroll
        for (int ks = 0; ks < 2; ++ks){
            v16bf af[4], bw[4];
            #pragma unroll
            for (int mi = 0; mi < 4; ++mi)
                af[mi] = frag_ld(&bufA[(wave_m * 64 + mi * 16 + l15) * TSTR], ks * 16 + half * 4);
            #pragma unroll
            for (int ni = 0; ni < 4; ++ni)
                bw[ni] = frag_ld(&bufB[(wave_n * 64 + ni * 16 + l15) * TSTR], ks * 16 + half * 4);
            #pragma unroll
            for (int mi = 0; mi < 4; ++mi)
                #pragma unroll
                for (int ni = 0; ni < 4; ++ni)
                    acc[mi][ni] = wmma_bf16(af[mi], bw[ni], acc[mi][ni]);
        }
    }

    // epilogue (all flags compile-time)
    #pragma unroll
    for (int mi = 0; mi < 4; ++mi){
        #pragma unroll
        for (int ni = 0; ni < 4; ++ni){
            const int row0 = bm + wave_m * 64 + mi * 16 + half * 8;
            const int col  = bn + wave_n * 64 + ni * 16 + l15;
            float bv = HAS_BIAS ? bias[col] : 0.0f;
            #pragma unroll
            for (int r = 0; r < 8; ++r){
                float v = acc[mi][ni][r] + bv;
                if (ACT == 1) v = gelu_f(v);
                v *= scale;
                if (OUT == 0){
                    ((float*)Cv)[(size_t)(row0 + r) * N + col] = v;
                } else if (OUT == 1){
                    ((u16*)Cv)[(size_t)(row0 + r) * N + col] = (u16)f2bf1(v);
                } else {
                    const int rr = row0 + r;   // token index
                    ((u16*)Cv)[((size_t)((rr >> 10) * H_ + col)) * S_ + (rr & (S_ - 1))] =
                        (u16)f2bf1(v);
                }
            }
        }
    }
}

// ---------------------------------------------------------------------------
// Flash attention (causal), all-bf16 inputs, async-double-buffered K/V tiles,
// one barrier per key-block.
// Q,K: bf16 [B*S][H] (Q pre-scaled by 1/8). Vt: bf16 [(b*H + h*64+d)][S].
// grid = (S/64, NH, B); block 128 (4 waves, 16 q-rows each). O: f32 [B*S][H].
// ---------------------------------------------------------------------------
#define KSTR 36

__global__ __launch_bounds__(128) void flash_attn_kernel(const u16* __restrict__ Q,
                                                         const u16* __restrict__ K,
                                                         const u16* __restrict__ Vt,
                                                         float* __restrict__ O){
    __shared__ __align__(16) u32 sK[2][64 * KSTR];   // [key][d]
    __shared__ __align__(16) u32 sV[2][64 * KSTR];   // [d][key]
    __shared__ __align__(16) u32 sP[4 * 16 * KSTR];  // per-wave probs [row][key]

    const int tid  = threadIdx.x;
    const int wave = tid >> 5, lane = tid & 31;
    const int half = lane >> 4, l15 = lane & 15;
    const int qb = blockIdx.x, hd = blockIdx.y, b = blockIdx.z;

    // Q fragments straight from global (bf16, pre-scaled)
    v16bf aq[2];
    {
        const u32* qp = (const u32*)(Q + (size_t)(b * S_ + qb * 64 + wave * 16 + l15) * H_ + hd * 64);
        aq[0] = frag_ld(qp, half * 4);
        aq[1] = frag_ld(qp, 16 + half * 4);
    }

    float mstate[8], lstate[8];
    #pragma unroll
    for (int r = 0; r < 8; ++r){ mstate[r] = -1e30f; lstate[r] = 0.0f; }
    v8f ctxa[4] = {};

    // staging: thread copies 64B of K-tile and 64B of V-tile per key-block
    const int cp_r = tid >> 1;   // 0..63 (key for sK, d for sV)
    const int cp_s = tid & 1;    // 32-bf16 half
    const u16* gK = K  + (size_t)(b * S_ + cp_r) * H_ + hd * 64 + cp_s * 32;  // + kb*64 rows
    const u16* gV = Vt + (size_t)(b * H_ + hd * 64 + cp_r) * S_ + cp_s * 32;  // + kb*64 cols
    u32 ldsK[2], ldsV[2];
    #pragma unroll
    for (int p = 0; p < 2; ++p){
        ldsK[p] = (u32)(size_t)&sK[p][cp_r * KSTR + cp_s * 16];
        ldsV[p] = (u32)(size_t)&sV[p][cp_r * KSTR + cp_s * 16];
    }

    async_copy64(ldsK[0], gK);
    async_copy64(ldsV[0], gV);

    for (int kb = 0; kb <= qb; ++kb){
        wait_async0();
        __syncthreads();
        if (kb < qb){
            async_copy64(ldsK[(kb + 1) & 1], gK + (size_t)(kb + 1) * 64 * H_);
            async_copy64(ldsV[(kb + 1) & 1], gV + (size_t)(kb + 1) * 64);
        }
        const u32* bufK = sK[kb & 1];
        const u32* bufV = sV[kb & 1];

        // scores = Q @ K^T (pre-scaled)
        v8f sc[4] = {};
        #pragma unroll
        for (int ks = 0; ks < 2; ++ks){
            #pragma unroll
            for (int nt = 0; nt < 4; ++nt){
                v16bf bk = frag_ld(&bufK[(nt * 16 + l15) * KSTR], ks * 16 + half * 4);
                sc[nt] = wmma_bf16(aq[ks], bk, sc[nt]);
            }
        }

        // causal mask + online softmax (rows live in vgpr index + lane-half)
        const int qrow0 = qb * 64 + wave * 16 + half * 8;
        #pragma unroll
        for (int nt = 0; nt < 4; ++nt){
            int col = kb * 64 + nt * 16 + l15;
            #pragma unroll
            for (int r = 0; r < 8; ++r)
                if (col > qrow0 + r) sc[nt][r] = -1e30f;
        }
        u16* sPb = (u16*)sP;
        #pragma unroll
        for (int r = 0; r < 8; ++r){
            float rm = fmaxf(fmaxf(sc[0][r], sc[1][r]), fmaxf(sc[2][r], sc[3][r]));
            rm = fmaxf(rm, __shfl_xor(rm, 1, 32));
            rm = fmaxf(rm, __shfl_xor(rm, 2, 32));
            rm = fmaxf(rm, __shfl_xor(rm, 4, 32));
            rm = fmaxf(rm, __shfl_xor(rm, 8, 32));
            float mnew  = fmaxf(mstate[r], rm);
            float alpha = __expf(mstate[r] - mnew);
            float psum  = 0.0f;
            #pragma unroll
            for (int nt = 0; nt < 4; ++nt){
                float p = __expf(sc[nt][r] - mnew);
                sc[nt][r] = p;
                psum += p;
            }
            psum += __shfl_xor(psum, 1, 32);
            psum += __shfl_xor(psum, 2, 32);
            psum += __shfl_xor(psum, 4, 32);
            psum += __shfl_xor(psum, 8, 32);
            lstate[r] = lstate[r] * alpha + psum;
            mstate[r] = mnew;
            #pragma unroll
            for (int dt = 0; dt < 4; ++dt)
                ctxa[dt][r] *= alpha;
            #pragma unroll
            for (int nt = 0; nt < 4; ++nt)
                sPb[(wave * 16 + half * 8 + r) * (2 * KSTR) + nt * 16 + l15] =
                    (u16)f2bf1(sc[nt][r]);
        }
        // sP is per-wave: wave-internal DS ordering (DScnt) covers the RAW.

        // ctx += probs @ V
        #pragma unroll
        for (int ks = 0; ks < 2; ++ks){
            v16bf ap = frag_ld(&sP[(wave * 16 + l15) * KSTR], ks * 16 + half * 4);
            #pragma unroll
            for (int dt = 0; dt < 4; ++dt){
                v16bf bv = frag_ld(&bufV[(dt * 16 + l15) * KSTR], ks * 16 + half * 4);
                ctxa[dt] = wmma_bf16(ap, bv, ctxa[dt]);
            }
        }
    }

    #pragma unroll
    for (int dt = 0; dt < 4; ++dt){
        #pragma unroll
        for (int r = 0; r < 8; ++r){
            int qrow = qb * 64 + wave * 16 + half * 8 + r;
            O[(size_t)(b * S_ + qrow) * H_ + hd * 64 + dt * 16 + l15] =
                ctxa[dt][r] / lstate[r];
        }
    }
}

// ---------------------------------------------------------------------------
// out = LayerNorm(x [+ y]) * g + b ; optional f32 and/or bf16 outputs.
// ---------------------------------------------------------------------------
template<int HAS_RES, int WF32, int WBF>
__global__ __launch_bounds__(256) void add_ln_kernel(const float* __restrict__ x,
                                                     const float* __restrict__ y,
                                                     const float* __restrict__ g,
                                                     const float* __restrict__ bb,
                                                     float* __restrict__ outf,
                                                     u16* __restrict__ outb){
    const int wave = threadIdx.x >> 5;
    const int lane = threadIdx.x & 31;
    const int row  = blockIdx.x * 8 + wave;

    const float* xp = x + (size_t)row * H_;
    const float* yp = HAS_RES ? y + (size_t)row * H_ : nullptr;

    float vals[32];
    float s = 0.0f, s2 = 0.0f;
    #pragma unroll
    for (int i = 0; i < 8; ++i){
        float4 a = *(const float4*)(xp + i * 128 + lane * 4);
        if (HAS_RES){
            float4 c = *(const float4*)(yp + i * 128 + lane * 4);
            a.x += c.x; a.y += c.y; a.z += c.z; a.w += c.w;
        }
        vals[i*4+0] = a.x; vals[i*4+1] = a.y; vals[i*4+2] = a.z; vals[i*4+3] = a.w;
        s  += a.x + a.y + a.z + a.w;
        s2 += a.x*a.x + a.y*a.y + a.z*a.z + a.w*a.w;
    }
    #pragma unroll
    for (int m = 1; m < 32; m <<= 1){
        s  += __shfl_xor(s,  m, 32);
        s2 += __shfl_xor(s2, m, 32);
    }
    const float inv  = 1.0f / (float)H_;
    float mean = s * inv;
    float rstd = rsqrtf(s2 * inv - mean * mean + 1e-5f);
    #pragma unroll
    for (int i = 0; i < 8; ++i){
        int col0 = i * 128 + lane * 4;
        float o[4];
        #pragma unroll
        for (int j = 0; j < 4; ++j)
            o[j] = (vals[i*4+j] - mean) * rstd * g[col0 + j] + bb[col0 + j];
        if (WF32){
            float4 f4; f4.x = o[0]; f4.y = o[1]; f4.z = o[2]; f4.w = o[3];
            *(float4*)(outf + (size_t)row * H_ + col0) = f4;
        }
        if (WBF){
            uint2 p; p.x = pack_bf2(o[0], o[1]); p.y = pack_bf2(o[2], o[3]);
            *(uint2*)(outb + (size_t)row * H_ + col0) = p;
        }
    }
}

// ---------------------------------------------------------------------------
extern "C" void kernel_launch(void* const* d_in, const int* in_sizes, int n_in,
                              void* d_out, int out_size, void* d_ws, size_t ws_size,
                              hipStream_t stream){
    (void)in_sizes; (void)n_in; (void)out_size; (void)ws_size;

    const int*   ids  = (const int*)  d_in[0];
    const float* emb  = (const float*)d_in[1];
    const float* pos  = (const float*)d_in[2];
    const float* Wq   = (const float*)d_in[3];
    const float* bq   = (const float*)d_in[4];
    const float* Wk   = (const float*)d_in[5];
    const float* bk   = (const float*)d_in[6];
    const float* Wv   = (const float*)d_in[7];
    const float* bv   = (const float*)d_in[8];
    const float* g1   = (const float*)d_in[9];
    const float* be1  = (const float*)d_in[10];
    const float* W1   = (const float*)d_in[11];
    const float* c1   = (const float*)d_in[12];
    const float* W2   = (const float*)d_in[13];
    const float* c2   = (const float*)d_in[14];
    const float* g2   = (const float*)d_in[15];
    const float* be2  = (const float*)d_in[16];
    const float* gf   = (const float*)d_in[17];
    const float* bf_  = (const float*)d_in[18];
    const float* Wlm  = (const float*)d_in[19];

    const size_t NT_ = (size_t)MR_ * H_;   // 2M elements
    char* w = (char*)d_ws;
    float* h    = (float*)w;            w += NT_ * 4;
    float* h1   = (float*)w;            w += NT_ * 4;
    float* ctx  = (float*)w;            w += NT_ * 4;
    u16*  hbf   = (u16*)w;              w += NT_ * 2;
    u16*  h1bf  = (u16*)w;              w += NT_ * 2;
    u16*  qbf   = (u16*)w;              w += NT_ * 2;
    u16*  kbf   = (u16*)w;              w += NT_ * 2;
    u16*  vt    = (u16*)w;              w += NT_ * 2;
    u16*  hfbf  = (u16*)w;              w += NT_ * 2;
    u16*  midbf = (u16*)w;              w += (size_t)MR_ * I_ * 2;
    u16*  warena= (u16*)w;              // up to H_*V_ bf16 (~64MB)

    u16* wqT = warena;                       // [H][H]
    u16* wkT = warena + (size_t)H_ * H_;
    u16* wvT = warena + (size_t)2 * H_ * H_;
    u16* w1T = warena + (size_t)3 * H_ * H_; // [I][H]
    u16* w2T = w1T    + (size_t)H_ * I_;     // [H][I]
    u16* wlmT= warena;                       // [V][H] (reuses arena after layers)

    dim3 blk(256);

    embed_kernel<<<(MR_ * H_) / 256, blk, 0, stream>>>(ids, emb, pos, h, hbf);

    for (int l = 0; l < L_; ++l){
        const float* wq = Wq + (size_t)l * H_ * H_;
        const float* wk = Wk + (size_t)l * H_ * H_;
        const float* wv = Wv + (size_t)l * H_ * H_;
        const float* w1 = W1 + (size_t)l * H_ * I_;
        const float* w2 = W2 + (size_t)l * I_ * H_;

        wt_kernel<<<dim3(H_/64, H_/64), blk, 0, stream>>>(wq, wqT, H_, H_);
        wt_kernel<<<dim3(H_/64, H_/64), blk, 0, stream>>>(wk, wkT, H_, H_);
        wt_kernel<<<dim3(H_/64, H_/64), blk, 0, stream>>>(wv, wvT, H_, H_);
        wt_kernel<<<dim3(H_/64, I_/64), blk, 0, stream>>>(w1, w1T, H_, I_);
        wt_kernel<<<dim3(I_/64, H_/64), blk, 0, stream>>>(w2, w2T, I_, H_);

        // QKV projections (bf16 out; Q pre-scaled by 1/sqrt(HD); V transposed)
        gemm_kernel<0,1,1><<<dim3(H_/256, MR_/128), blk, 0, stream>>>(
            hbf, wqT, bq + (size_t)l * H_, qbf, MR_, H_, H_, 0.125f);
        gemm_kernel<0,1,1><<<dim3(H_/256, MR_/128), blk, 0, stream>>>(
            hbf, wkT, bk + (size_t)l * H_, kbf, MR_, H_, H_, 1.0f);
        gemm_kernel<0,1,2><<<dim3(H_/256, MR_/128), blk, 0, stream>>>(
            hbf, wvT, bv + (size_t)l * H_, vt, MR_, H_, H_, 1.0f);

        flash_attn_kernel<<<dim3(S_/64, NH_, B_), dim3(128), 0, stream>>>(qbf, kbf, vt, ctx);

        add_ln_kernel<1,1,1><<<MR_/8, blk, 0, stream>>>(
            h, ctx, g1 + (size_t)l * H_, be1 + (size_t)l * H_, h1, h1bf);

        gemm_kernel<1,1,1><<<dim3(I_/256, MR_/128), blk, 0, stream>>>(
            h1bf, w1T, c1 + (size_t)l * I_, midbf, MR_, I_, H_, 1.0f);
        gemm_kernel<0,1,0><<<dim3(H_/256, MR_/128), blk, 0, stream>>>(
            midbf, w2T, c2 + (size_t)l * H_, ctx, MR_, H_, I_, 1.0f);

        add_ln_kernel<1,1,1><<<MR_/8, blk, 0, stream>>>(
            h1, ctx, g2 + (size_t)l * H_, be2 + (size_t)l * H_, h, hbf);
    }

    add_ln_kernel<0,0,1><<<MR_/8, blk, 0, stream>>>(h, nullptr, gf, bf_, nullptr, hfbf);
    wt_kernel<<<dim3(H_/64, V_/64), blk, 0, stream>>>(Wlm, wlmT, H_, V_);
    gemm_kernel<0,0,0><<<dim3(V_/256, MR_/128), blk, 0, stream>>>(
        hfbf, wlmT, nullptr, (float*)d_out, MR_, V_, H_, 1.0f);
}
